// GeoGNN_87823491268924
// MI455X (gfx1250) — matrix operand order repaired and checked by reading, not compiled, not run on hardware
//
#include <hip/hip_runtime.h>
#include <math.h>

// ============================================================================
// GeoGNN forward for MI455X (gfx1250, wave32, WMMA + Tensor Data Mover).
// Core GEMM: v_wmma_f32_16x16x32_f16, f32 accumulate. Weights pre-converted
// once to f16 *transposed* [N][Kpad] so the per-K-step B panel is a plain 2D
// tile that the TDM (tensor_load_to_lds) DMAs into LDS, double-buffered;
// fragments are read as 2x ds_load_b128. A rows are read exactly once per
// GEMM as float4 loads. Fused bias+activation epilogue.
// ============================================================================

#define NN      10000      // nodes
#define NE      250000     // edges
#define NG      32         // graphs
#define HID     128
#define NHEAD   4
#define HDIM    32
#define HV      1024
#define NIN     1208       // node input features
#define NINP    1216       // padded to mult of 32
#define EIN     450        // edge input features
#define EINP    480        // padded to mult of 32
#define ECH     31250      // edge chunk rows (8 chunks of 31250)
#define BNB     240        // batchnorm partial blocks

typedef __attribute__((ext_vector_type(16))) _Float16 v16h;
typedef __attribute__((ext_vector_type(8)))  float    v8f;
typedef __attribute__((ext_vector_type(4)))  unsigned u32x4;
typedef __attribute__((ext_vector_type(4)))  int      i32x4;
typedef __attribute__((ext_vector_type(8)))  int      i32x8;

union AFrag { v16h v; _Float16 h[16]; uint4 u[2]; };
union CFrag { v8f  v; float    f[8];  };

#if defined(__has_builtin)
#if __has_builtin(__builtin_amdgcn_tensor_load_to_lds) && \
    __has_builtin(__builtin_amdgcn_s_wait_tensorcnt)
#define USE_TDM 1
#endif
#endif
#ifndef USE_TDM
#define USE_TDM 0
#endif

// ---------------------------------------------------------------- helpers ---
__device__ __forceinline__ float d3(const float* a, const float* b) {
    return a[0]*b[0] + a[1]*b[1] + a[2]*b[2];
}
__device__ __forceinline__ void c3(const float* a, const float* b, float* o) {
    o[0] = a[1]*b[2] - a[2]*b[1];
    o[1] = a[2]*b[0] - a[0]*b[2];
    o[2] = a[0]*b[1] - a[1]*b[0];
}
__device__ __forceinline__ void n3(float* v) {
    float n = fmaxf(sqrtf(d3(v, v)), 1e-12f);
    v[0] /= n; v[1] /= n; v[2] /= n;
}
__device__ __forceinline__ float sgnf(float x) {
    return (x > 0.f ? 1.f : 0.f) - (x < 0.f ? 1.f : 0.f);
}
__device__ __forceinline__ float clampf(float x, float lo, float hi) {
    return fminf(fmaxf(x, lo), hi);
}
__device__ __forceinline__ float act_apply(float x, int a) {
    switch (a) {
        case 1: return fmaxf(x, 0.f);                              // relu
        case 2: return 0.5f * x * (1.f + erff(x * 0.70710678f));   // gelu exact
        case 3: return 1.f / (1.f + expf(-x));                     // sigmoid
        case 4: return x > 0.f ? x : expm1f(x);                    // elu
    }
    return x;
}
// order-independent float max via monotonic uint encoding
__device__ __forceinline__ unsigned encf(float x) {
    int i = __float_as_int(x);
    return (i >= 0) ? ((unsigned)i | 0x80000000u) : ~((unsigned)i);
}
__device__ __forceinline__ float decf(unsigned u) {
    int i = (u & 0x80000000u) ? (int)(u & 0x7FFFFFFFu) : (int)(~u);
    return __int_as_float(i);
}

#if USE_TDM
// Issue a TDM load of a [128 rows x 32 halfs] f16 tile (row stride Kpad halfs)
// from global gptr into LDS at byte offset ldsOff. D# per CDNA5 ISA ch.8:
// group0: count=1 | lds_addr | global_addr(57b) | type=2("image")
// group1: data_size=1(2B), tensor_dim0=32, tensor_dim1=128,
//         tile_dim0=32, tile_dim1=128, tile_dim2=0, dim0_stride=Kpad
__device__ __forceinline__ void tdm_issue(const _Float16* gptr, unsigned ldsOff, int Kpad)
{
    unsigned long long ga = (unsigned long long)(size_t)gptr;
    u32x4 g0 = { 1u,                                   // count=1, user mode
                 ldsOff,                               // lds_addr (bytes)
                 (unsigned)ga,                         // global_addr[31:0]
                 (unsigned)((ga >> 32) & 0x01FFFFFFu) | 0x80000000u }; // [56:32] | type=2
    i32x8 g1 = { 0x00010000,                           // data_size=1 (2 bytes)
                 (int)(32u << 16),                     // tensor_dim0[15:0]=32
                 (int)(128u << 16),                    // dim0 hi=0 | tensor_dim1[15:0]=128
                 (int)(32u << 16),                     // dim1 hi=0 | tile_dim0=32
                 128,                                  // tile_dim1=128, tile_dim2=0
                 Kpad,                                 // tensor_dim0_stride[31:0]
                 0, 0 };                               // stride hi, dim1_stride=0
    i32x4 z4 = { 0, 0, 0, 0 };
#if __clang_major__ >= 23
    i32x8 z8 = { 0, 0, 0, 0, 0, 0, 0, 0 };
    __builtin_amdgcn_tensor_load_to_lds(g0, g1, z4, z4, z8, 0);
#else
    __builtin_amdgcn_tensor_load_to_lds(g0, g1, z4, z4, 0);
#endif
}
#endif

// ============================================== weight convert/transpose ====
// Bt[n*Kpad + k] = (f16)B[k*N + n], zero-padded for k >= K
__global__ void k_wcvt(const float* __restrict__ B, _Float16* __restrict__ Bt,
                       int K, int Kpad, int N)
{
    size_t total = (size_t)N * Kpad;
    for (size_t i = (size_t)blockIdx.x * blockDim.x + threadIdx.x; i < total;
         i += (size_t)gridDim.x * blockDim.x) {
        int n = (int)(i / Kpad), k = (int)(i % Kpad);
        Bt[i] = (k < K) ? (_Float16)B[(size_t)k * N + n] : (_Float16)0.f;
    }
}

// =========================================================== WMMA GEMM ======
// C[M x N] = act(A[M x K] * W + bias); W given as f16 transposed Bt[N][Kpad].
// Block: 256 thr = 8 waves; block tile 128(M) x 128(N); grid.y = N/128 panels.
// B panel [128 n][32 k] staged in LDS by TDM (double-buffered) or coop copy.
__launch_bounds__(256)
__global__ void k_gemm(const float* __restrict__ A, const _Float16* __restrict__ Bt,
                       const float* __restrict__ bias, float* __restrict__ C,
                       int M, int Kpad, int N, int lda, int ldc, int act)
{
    __shared__ __align__(16) _Float16 panel[2][128 * 32];   // 2 x 8KB

    const int tid  = threadIdx.x;
    const int lane = tid & 31;
    const int wave = tid >> 5;
    const int m0   = blockIdx.x * 128 + wave * 16;
    const int nbase = blockIdx.y * 128;
    const int mrow = m0 + (lane & 15);
    const bool mvalid = mrow < M;
    const _Float16* btBase = Bt + (size_t)nbase * Kpad;

    CFrag acc[8];
#pragma unroll
    for (int t = 0; t < 8; t++)
#pragma unroll
        for (int r = 0; r < 8; r++) acc[t].f[r] = 0.f;

#if USE_TDM
    if (tid < 32) tdm_issue(btBase, (unsigned)(size_t)&panel[0][0], Kpad);
#endif

    int cur = 0;
    for (int k0 = 0; k0 < Kpad; k0 += 32) {
#if USE_TDM
        if (tid < 32) __builtin_amdgcn_s_wait_tensorcnt(0);
        __syncthreads();                         // panel[cur] ready, prev reads done
        if (tid < 32 && k0 + 32 < Kpad)
            tdm_issue(btBase + (k0 + 32), (unsigned)(size_t)&panel[cur ^ 1][0], Kpad);
#else
        __syncthreads();
        for (int i = tid; i < 512; i += 256) {   // 128 rows x 4 uint4
            int row = i >> 2, q = i & 3;
            ((uint4*)&panel[cur][row * 32])[q] =
                ((const uint4*)(btBase + (size_t)row * Kpad + k0))[q];
        }
        __syncthreads();
#endif
        // A fragment (16x32 f16): lane&15 = row, lane>>4 selects K-halves
        AFrag af;
        {
            const int kb = k0 + ((lane >> 4) << 3);
            const float4* ap = (const float4*)(A + (size_t)mrow * lda + kb);
            float av[16];
            if (mvalid) {
                *(float4*)&av[0]  = ap[0];
                *(float4*)&av[4]  = ap[1];
                *(float4*)&av[8]  = ap[4];       // +16 floats
                *(float4*)&av[12] = ap[5];
                __builtin_prefetch((const float*)ap + 32, 0, 1); // global_prefetch_b8
            } else {
#pragma unroll
                for (int j = 0; j < 16; j++) av[j] = 0.f;
            }
#pragma unroll
            for (int j = 0; j < 16; j++) af.h[j] = (_Float16)av[j];
        }

        const _Float16* pc = &panel[cur][0];
#pragma unroll
        for (int nt = 0; nt < 8; nt++) {
            AFrag bf;
            const int nl = nt * 16 + (lane & 15);
            const int ko = (lane >> 4) << 3;
            const uint4* bp = (const uint4*)(pc + nl * 32 + ko);   // 16B aligned
            bf.u[0] = bp[0];
            bf.u[1] = bp[2];
            acc[nt].v = __builtin_amdgcn_wmma_f32_16x16x32_f16(
                false, af.v, false, bf.v, (short)0, acc[nt].v, false, false);
        }
        cur ^= 1;
    }

    // epilogue: VGPR r -> row r (lanes 0-15) / 8+r (lanes 16-31), col = lane&15
    const int rbase = (lane >> 4) << 3;
#pragma unroll
    for (int nt = 0; nt < 8; nt++) {
        const int n = nbase + nt * 16 + (lane & 15);
        const float bb = bias ? bias[n] : 0.f;
#pragma unroll
        for (int r = 0; r < 8; r++) {
            const int m = m0 + rbase + r;
            if (m < M) {
                float v = act_apply(acc[nt].f[r] + bb, act);
                C[(size_t)m * ldc + n] = v;
            }
        }
    }
}

// ===================================================== geometry kernels =====
// concat buffer layout per node (NINP cols): [0..1023]=h_V, [1024..1035]=angles,
// [1036..1195]=node_dist, [1196..1207]=node_dir, [1208..1215]=0
__global__ void k_copy_hv(const float* __restrict__ hv, float* __restrict__ nf)
{
    size_t total = (size_t)NN * NINP;
    for (size_t i = (size_t)blockIdx.x * blockDim.x + threadIdx.x; i < total;
         i += (size_t)gridDim.x * blockDim.x) {
        size_t n = i / NINP; int c = (int)(i % NINP);
        if (c < HV)        nf[i] = hv[n * HV + c];
        else if (c >= NIN) nf[i] = 0.f;
    }
}

__global__ void k_angles(const float* __restrict__ X, float* __restrict__ nf)
{
    int j = blockIdx.x * blockDim.x + threadIdx.x;
    if (j >= 3 * NN) return;
    float Dv = 0.f, Bv = 0.f;
    if (j >= 1 && j <= 3 * NN - 3) {
        float P[4][3];
#pragma unroll
        for (int t = 0; t < 4; t++) {
            int idx = j - 1 + t, n = idx / 3, a = idx % 3;
            for (int d = 0; d < 3; d++) P[t][d] = X[n * 15 + a * 3 + d];
        }
        float u2[3], u1[3], u0[3], nv2[3], nv1[3];
        for (int d = 0; d < 3; d++) { u2[d] = P[1][d] - P[0][d]; u1[d] = P[2][d] - P[1][d]; u0[d] = P[3][d] - P[2][d]; }
        n3(u2); n3(u1); n3(u0);
        c3(u2, u1, nv2); n3(nv2);
        c3(u1, u0, nv1); n3(nv1);
        float cosD = clampf(d3(nv2, nv1), -1.f + 1e-7f, 1.f - 1e-7f);
        Dv = sgnf(d3(u2, nv1)) * acosf(cosD);
        Bv = acosf(clampf(d3(u2, u1), -1.f + 1e-7f, 1.f - 1e-7f));
    }
    int node = j / 3, t = j % 3;
    float* g = nf + (size_t)node * NINP + HV;
    g[t] = cosf(Dv); g[3 + t] = sinf(Dv); g[6 + t] = cosf(Bv); g[9 + t] = sinf(Bv);
}

__global__ void k_node_geo(const float* __restrict__ X, float* __restrict__ nf,
                           float* __restrict__ lfb)
{
    int n = blockIdx.x * blockDim.x + threadIdx.x;
    if (n >= NN) return;
    float x[5][3];
    for (int a = 0; a < 5; a++)
        for (int d = 0; d < 3; d++) x[a][d] = X[n * 15 + a * 3 + d];
    float* g = nf + (size_t)n * NINP + HV;
    const int p0[10] = {1, 1, 1, 0, 0, 3, 4, 4, 4, 4};
    const int p1[10] = {0, 2, 3, 2, 3, 2, 0, 1, 2, 3};
    for (int p = 0; p < 10; p++) {
        float dx = x[p0[p]][0] - x[p1[p]][0], dy = x[p0[p]][1] - x[p1[p]][1], dz = x[p0[p]][2] - x[p1[p]][2];
        float D = sqrtf(dx*dx + dy*dy + dz*dz);
        for (int j = 0; j < 16; j++) {
            float t = (D - j * (20.f / 15.f)) * 0.8f;   // sigma = 1.25
            g[12 + p * 16 + j] = expf(-t * t);
        }
    }
    // local frame lf[d][k]: col0=b, col1=nn, col2=b x nn
    float u[3], v[3], b[3], nr[3], cc[3];
    for (int d = 0; d < 3; d++) { u[d] = x[1][d] - x[0][d]; v[d] = x[2][d] - x[1][d]; }
    n3(u); n3(v);
    for (int d = 0; d < 3; d++) b[d] = u[d] - v[d];
    n3(b);
    c3(u, v, nr); n3(nr);
    c3(b, nr, cc);
    for (int d = 0; d < 3; d++) {
        lfb[(size_t)n * 9 + d * 3 + 0] = b[d];
        lfb[(size_t)n * 9 + d * 3 + 1] = nr[d];
        lfb[(size_t)n * 9 + d * 3 + 2] = cc[d];
    }
    const int aidx[4] = {0, 2, 3, 4};
    for (int ai = 0; ai < 4; ai++) {
        float t[3];
        for (int d = 0; d < 3; d++) t[d] = x[aidx[ai]][d] - x[1][d];
        n3(t);
        for (int k = 0; k < 3; k++)
            g[172 + ai * 3 + k] = t[0]*lfb[(size_t)n*9 + 0*3 + k] + t[1]*lfb[(size_t)n*9 + 1*3 + k] + t[2]*lfb[(size_t)n*9 + 2*3 + k];
    }
}

// geo_edge per row (EINP cols): [0..15]=pos, [16..19]=quat, [20..419]=dist,
// [420..449]=dir, [450..479]=0
__global__ void k_edge_geo(const float* __restrict__ X, const float* __restrict__ lfb,
                           const int* __restrict__ src, const int* __restrict__ dst,
                           int e0, int cnt, float* __restrict__ ge)
{
    int e = blockIdx.x * blockDim.x + threadIdx.x;
    if (e >= cnt) return;
    int g = e0 + e;
    int s = src[g], t = dst[g];
    float* o = ge + (size_t)e * EINP;

    // positional embedding
    float dd = (float)(s - t);
    for (int k = 0; k < 8; k++) {
        float fr = expf((float)(2 * k) * (-9.2103403719f / 16.f));  // ln(1e4)
        float ang = dd * fr;
        o[k] = cosf(ang); o[8 + k] = sinf(ang);
    }
    float xs[5][3], xd[5][3], lfi[9], lfj[9];
    for (int a = 0; a < 5; a++)
        for (int d = 0; d < 3; d++) { xs[a][d] = X[s * 15 + a * 3 + d]; xd[a][d] = X[t * 15 + a * 3 + d]; }
    for (int i = 0; i < 9; i++) { lfi[i] = lfb[(size_t)t * 9 + i]; lfj[i] = lfb[(size_t)s * 9 + i]; }

    // quaternion of R = lf_i^T lf_j
    float R[3][3];
    for (int k = 0; k < 3; k++)
        for (int m = 0; m < 3; m++)
            R[k][m] = lfi[0*3+k]*lfj[0*3+m] + lfi[1*3+k]*lfj[1*3+m] + lfi[2*3+k]*lfj[2*3+m];
    float Rxx = R[0][0], Ryy = R[1][1], Rzz = R[2][2];
    float q0 = sgnf(R[2][1]-R[1][2]) * 0.5f * sqrtf(fabsf(1.f + Rxx - Ryy - Rzz));
    float q1 = sgnf(R[0][2]-R[2][0]) * 0.5f * sqrtf(fabsf(1.f - Rxx + Ryy - Rzz));
    float q2 = sgnf(R[1][0]-R[0][1]) * 0.5f * sqrtf(fabsf(1.f - Rxx - Ryy + Rzz));
    float q3 = 0.5f * sqrtf(fmaxf(0.f, 1.f + Rxx + Ryy + Rzz));
    float qn = fmaxf(sqrtf(q0*q0 + q1*q1 + q2*q2 + q3*q3), 1e-12f);
    o[16] = q0/qn; o[17] = q1/qn; o[18] = q2/qn; o[19] = q3/qn;

    // edge rbf distances (5x5x16)
    for (int a = 0; a < 5; a++)
        for (int b = 0; b < 5; b++) {
            float dx = xs[a][0]-xd[b][0], dy = xs[a][1]-xd[b][1], dz = xs[a][2]-xd[b][2];
            float D = sqrtf(dx*dx + dy*dy + dz*dz);
            for (int j = 0; j < 16; j++) {
                float tt = (D - j * (20.f / 15.f)) * 0.8f;
                o[20 + (a * 5 + b) * 16 + j] = expf(-tt * tt);
            }
        }
    // directions
    for (int a = 0; a < 5; a++) {
        float tv[3];
        for (int d = 0; d < 3; d++) tv[d] = xs[a][d] - xd[1][d];
        n3(tv);
        for (int k = 0; k < 3; k++)
            o[420 + a * 3 + k] = tv[0]*lfi[0*3+k] + tv[1]*lfi[1*3+k] + tv[2]*lfi[2*3+k];
    }
    for (int a = 0; a < 5; a++) {
        float tv[3];
        for (int d = 0; d < 3; d++) tv[d] = xd[a][d] - xs[1][d];
        n3(tv);
        for (int k = 0; k < 3; k++)
            o[435 + a * 3 + k] = tv[0]*lfj[0*3+k] + tv[1]*lfj[1*3+k] + tv[2]*lfj[2*3+k];
    }
    for (int c = EIN; c < EINP; c++) o[c] = 0.f;
}

// ======================================================= norm kernels =======
__global__ void k_bn_part(const float* __restrict__ x, int M, float* __restrict__ part)
{
    int c = threadIdx.x;                    // 128 threads
    float s = 0.f, ss = 0.f;
    for (int row = blockIdx.x; row < M; row += gridDim.x) {
        float v = x[(size_t)row * HID + c];
        s += v; ss += v * v;
    }
    part[(blockIdx.x * 2 + 0) * HID + c] = s;
    part[(blockIdx.x * 2 + 1) * HID + c] = ss;
}
__global__ void k_bn_final(const float* __restrict__ part, int nb, int M,
                           float* __restrict__ mu, float* __restrict__ istd)
{
    int c = threadIdx.x;
    float s = 0.f, ss = 0.f;
    for (int b = 0; b < nb; b++) { s += part[(b*2+0)*HID + c]; ss += part[(b*2+1)*HID + c]; }
    float m = s / (float)M;
    float v = ss / (float)M - m * m;
    mu[c] = m;
    istd[c] = rsqrtf(fmaxf(v, 0.f) + 1e-5f);
}
__global__ void k_bn_apply(const float* __restrict__ x, const float* __restrict__ mu,
                           const float* __restrict__ istd, const float* __restrict__ g,
                           const float* __restrict__ b, float* __restrict__ y, size_t total)
{
    for (size_t i = (size_t)blockIdx.x * blockDim.x + threadIdx.x; i < total;
         i += (size_t)gridDim.x * blockDim.x) {
        int c = (int)(i & (HID - 1));
        y[i] = (x[i] - mu[c]) * istd[c] * g[c] + b[c];
    }
}
__global__ void k_layernorm(const float* __restrict__ x, const float* __restrict__ r,
                            const float* __restrict__ g, const float* __restrict__ b,
                            float* __restrict__ y)
{
    __shared__ float sh[HID];
    int row = blockIdx.x, c = threadIdx.x;
    float v = x[(size_t)row * HID + c] + (r ? r[(size_t)row * HID + c] : 0.f);
    sh[c] = v; __syncthreads();
    for (int s = 64; s > 0; s >>= 1) { if (c < s) sh[c] += sh[c + s]; __syncthreads(); }
    float mu = sh[0] / (float)HID; __syncthreads();
    float d = v - mu;
    sh[c] = d * d; __syncthreads();
    for (int s = 64; s > 0; s >>= 1) { if (c < s) sh[c] += sh[c + s]; __syncthreads(); }
    float var = sh[0] / (float)HID;
    y[(size_t)row * HID + c] = d * rsqrtf(var + 1e-5f) * g[c] + b[c];
}

// =================================================== attention kernels ======
__global__ void k_attn_logits(const float* __restrict__ q, const float* __restrict__ k,
                              const float* __restrict__ ef, const int* __restrict__ src,
                              const int* __restrict__ dst, float* __restrict__ alpha)
{
    int i = blockIdx.x * blockDim.x + threadIdx.x;
    if (i >= NE * NHEAD) return;
    int e = i >> 2, h = i & 3;
    size_t qs = (size_t)dst[e] * HID + h * HDIM;
    size_t ks = (size_t)src[e] * HID + h * HDIM;
    size_t es = (size_t)e * HID + h * HDIM;
    float s = 0.f;
#pragma unroll 8
    for (int d = 0; d < HDIM; d++) s += q[qs + d] * (k[ks + d] + ef[es + d]);
    alpha[i] = s * 0.1767766953f;   // 1/sqrt(32)
}
__global__ void k_fill_u32(unsigned* p, size_t n, unsigned v)
{
    for (size_t i = (size_t)blockIdx.x * blockDim.x + threadIdx.x; i < n;
         i += (size_t)gridDim.x * blockDim.x) p[i] = v;
}
__global__ void k_seg_max(const float* __restrict__ alpha, const int* __restrict__ dst,
                          unsigned* __restrict__ mEnc)
{
    int i = blockIdx.x * blockDim.x + threadIdx.x;
    if (i >= NE * NHEAD) return;
    int e = i >> 2, h = i & 3;
    atomicMax(&mEnc[dst[e] * NHEAD + h], encf(alpha[i]));
}
__global__ void k_seg_expsum(float* __restrict__ alpha, const int* __restrict__ dst,
                             const unsigned* __restrict__ mEnc, float* __restrict__ ssum)
{
    int i = blockIdx.x * blockDim.x + threadIdx.x;
    if (i >= NE * NHEAD) return;
    int e = i >> 2, h = i & 3;
    float m = decf(mEnc[dst[e] * NHEAD + h]);
    float ex = expf(alpha[i] - m);
    alpha[i] = ex;                      // store numerator in place
    atomicAdd(&ssum[dst[e] * NHEAD + h], ex);
}
__global__ void k_attn_scatter(const float* __restrict__ vn, const float* __restrict__ ef,
                               const float* __restrict__ ex, const float* __restrict__ ssum,
                               const int* __restrict__ src, const int* __restrict__ dst,
                               float* __restrict__ dh)
{
    int e = blockIdx.x, c = threadIdx.x;   // 128 threads / edge
    int h = c >> 5;
    int dn = dst[e], sn = src[e];
    float w = ex[e * NHEAD + h] / (ssum[dn * NHEAD + h] + 1e-16f);
    float val = (vn[(size_t)sn * HID + c] + ef[(size_t)e * HID + c]) * w;
    atomicAdd(&dh[(size_t)dn * HID + c], val);
}

// ===================================================== misc kernels =========
__global__ void k_add(const float* a, const float* b, float* o, size_t n)
{
    for (size_t i = (size_t)blockIdx.x * blockDim.x + threadIdx.x; i < n;
         i += (size_t)gridDim.x * blockDim.x) o[i] = a[i] + b[i];
}
__global__ void k_concat_hEV(const float* __restrict__ hv, const float* __restrict__ hE,
                             const int* __restrict__ src, const int* __restrict__ dst,
                             int e0, int cnt, float* __restrict__ hEV)
{
    size_t total = (size_t)cnt * 384;
    for (size_t i = (size_t)blockIdx.x * blockDim.x + threadIdx.x; i < total;
         i += (size_t)gridDim.x * blockDim.x) {
        size_t e = i / 384; int c = (int)(i % 384);
        int g = e0 + (int)e;
        float v;
        if (c < 128)      v = hv[(size_t)src[g] * HID + c];
        else if (c < 256) v = hE[(size_t)g * HID + (c - 128)];
        else              v = hv[(size_t)dst[g] * HID + (c - 256)];
        hEV[(size_t)e * 384 + c] = v;
    }
}
__global__ void k_counts(const int* __restrict__ batch, int* __restrict__ cnt)
{
    int i = blockIdx.x * blockDim.x + threadIdx.x;
    if (i < NN) atomicAdd(&cnt[batch[i]], 1);
}
__global__ void k_seg_sum(const float* __restrict__ x, const int* __restrict__ batch,
                          float* __restrict__ acc)
{
    size_t total = (size_t)NN * HID;
    for (size_t i = (size_t)blockIdx.x * blockDim.x + threadIdx.x; i < total;
         i += (size_t)gridDim.x * blockDim.x) {
        size_t n = i >> 7; int c = (int)(i & 127);
        atomicAdd(&acc[(size_t)batch[n] * HID + c], x[i]);
    }
}
__global__ void k_div_cnt(const float* __restrict__ acc, const int* __restrict__ cnt,
                          float* __restrict__ out)
{
    int i = blockIdx.x * blockDim.x + threadIdx.x;
    if (i >= NG * HID) return;
    out[i] = acc[i] / fmaxf((float)cnt[i >> 7], 1.f);
}
__global__ void k_gate_apply(float* __restrict__ hv, const float* __restrict__ gate,
                             const int* __restrict__ batch)
{
    size_t total = (size_t)NN * HID;
    for (size_t i = (size_t)blockIdx.x * blockDim.x + threadIdx.x; i < total;
         i += (size_t)gridDim.x * blockDim.x) {
        size_t n = i >> 7; int c = (int)(i & 127);
        hv[i] *= gate[(size_t)batch[n] * HID + c];
    }
}

// ============================================================ host side =====
extern "C" void kernel_launch(void* const* d_in, const int* in_sizes, int n_in,
                              void* d_out, int out_size, void* d_ws, size_t ws_size,
                              hipStream_t stream)
{
    if (n_in < 118) return;
    // d_in layout (JAX pytree sorted-key flattening):
    // 0:X 1:h_V 2:We_W 3:We_b 4:Wv_W 5:Wv_b 6:bn_e_b 7:bn_e_g 8:bn_n_b 9:bn_n_g
    // 10:edge_emb_W 11:edge_emb_b 12:fc_W 13:fc_b 14..113:layers(4x25)
    // 114:node_emb_W 115:node_emb_b 116:edge_index 117:batch_id
    const float* X   = (const float*)d_in[0];
    const float* hV0 = (const float*)d_in[1];
    const int* src   = (const int*)d_in[116];
    const int* dst   = ((const int*)d_in[116]) + NE;
    const int* batch = (const int*)d_in[117];
    auto PP = [&](int i) { return (const float*)d_in[i]; };
    auto LP = [&](int L, int w) { return (const float*)d_in[14 + 25 * L + w]; };
    // layer leaf offsets (sorted): 0 W11,1 W12,2 We_a,3 Wk,4 Wq,5 Wv_a,6 b11,7 b12,
    // 8 bk,9 bn_b,10 bn_g,11 bq,12 bv_a,13 ctx_W1,14 ctx_W2,15 ctx_b1,16 ctx_b2,
    // 17 ffn_W1,18 ffn_W2,19 ffn_b1,20 ffn_b2,21 ln0_b,22 ln0_g,23 ln1_b,24 ln1_g

    // ---- workspace allocator ----
    float* w = (float*)d_ws;
    size_t off = 0;
    auto alloc = [&](size_t nf) { float* p = w + off; off += (nf + 63) & ~(size_t)63; return p; };
    float* nf    = alloc((size_t)NN * NINP);
    float* lfb   = alloc((size_t)NN * 9);
    float* hv_a  = alloc((size_t)NN * HID);
    float* hv_b  = alloc((size_t)NN * HID);
    float* dh    = alloc((size_t)NN * HID);
    float* qn    = alloc((size_t)NN * HID);
    float* kn    = alloc((size_t)NN * HID);
    float* vn    = alloc((size_t)NN * HID);
    float* nb1   = alloc((size_t)NN * HID);
    float* t1    = alloc((size_t)NN * 512);
    float* mu    = alloc(HID);
    float* istd  = alloc(HID);
    float* part  = alloc((size_t)BNB * 2 * HID);
    float* mEncF = alloc((size_t)NN * NHEAD);
    float* ssum  = alloc((size_t)NN * NHEAD);
    float* alpha = alloc((size_t)NE * NHEAD);
    float* csum  = alloc((size_t)NG * HID);
    float* cbuf  = alloc((size_t)NG * HID);
    float* g1    = alloc((size_t)NG * HID);
    float* gate  = alloc((size_t)NG * HID);
    float* cntF  = alloc(64);
    float* E1    = alloc((size_t)NE * HID);   // scratch / sums
    float* E2    = alloc((size_t)NE * HID);   // h_E
    float* E3    = alloc((size_t)NE * HID);   // e_feat / msg2
    float* E4    = alloc((size_t)NE * HID);   // msg1
    float* geoE  = alloc((size_t)ECH * EINP);
    float* hEV   = alloc((size_t)ECH * 384);
    unsigned* mEnc = (unsigned*)mEncF;
    int* cnt = (int*)cntF;

    auto gs = [](size_t n) { return dim3((unsigned)((n + 255) / 256)); };
    // f16 transposed-weight allocator + converter
    auto wcvt = [&](const float* B, int K, int Kpad, int N) {
        _Float16* p = (_Float16*)alloc(((size_t)N * Kpad + 1) / 2);
        k_wcvt<<<gs((size_t)N * Kpad), dim3(256), 0, stream>>>(B, p, K, Kpad, N);
        return p;
    };
    // pre-convert all weights (tiny: ~2.6MB total, once per launch)
    _Float16* bt_ne = wcvt(PP(114), NIN, NINP, HID);
    _Float16* bt_wv = wcvt(PP(4),  HID, HID, HID);
    _Float16* bt_we = wcvt(PP(2),  HID, HID, HID);
    _Float16* bt_ee = wcvt(PP(10), EIN, EINP, HID);
    _Float16* bt_fc = wcvt(PP(12), HID, HID, HID);
    _Float16* btL[4][10];
    for (int L = 0; L < 4; L++) {
        btL[L][0] = wcvt(LP(L, 4),  HID, HID, HID);   // Wq
        btL[L][1] = wcvt(LP(L, 3),  HID, HID, HID);   // Wk
        btL[L][2] = wcvt(LP(L, 5),  HID, HID, HID);   // Wv_a
        btL[L][3] = wcvt(LP(L, 2),  HID, HID, HID);   // We_a
        btL[L][4] = wcvt(LP(L, 17), HID, HID, 512);   // ffn_W1
        btL[L][5] = wcvt(LP(L, 18), 512, 512, HID);   // ffn_W2
        btL[L][6] = wcvt(LP(L, 0),  384, 384, HID);   // W11
        btL[L][7] = wcvt(LP(L, 1),  HID, HID, HID);   // W12
        btL[L][8] = wcvt(LP(L, 13), HID, HID, HID);   // ctx_W1
        btL[L][9] = wcvt(LP(L, 14), HID, HID, HID);   // ctx_W2
    }
    if (off * sizeof(float) > ws_size) return;   // workspace too small

    auto gemm = [&](const float* A, const _Float16* Bt, const float* bias, float* C,
                    int M, int Kpad, int N, int lda, int ldc, int act) {
        dim3 grid((unsigned)((M + 127) / 128), (unsigned)(N / 128));
        k_gemm<<<grid, dim3(256), 0, stream>>>(A, Bt, bias, C, M, Kpad, N, lda, ldc, act);
    };
    auto bn = [&](const float* x, const float* gg, const float* bb, float* y, int M) {
        k_bn_part<<<dim3(BNB), dim3(HID), 0, stream>>>(x, M, part);
        k_bn_final<<<dim3(1), dim3(HID), 0, stream>>>(part, BNB, M, mu, istd);
        size_t tot = (size_t)M * HID;
        k_bn_apply<<<dim3((unsigned)((tot + 255) / 256)), dim3(256), 0, stream>>>(x, mu, istd, gg, bb, y, tot);
    };

    // ---- graph counts ----
    hipMemsetAsync(cnt, 0, NG * sizeof(int), stream);
    k_counts<<<gs(NN), dim3(256), 0, stream>>>(batch, cnt);

    // ---- node features + embedding ----
    k_copy_hv<<<gs((size_t)NN * NINP), dim3(256), 0, stream>>>(hV0, nf);
    k_angles<<<gs(3 * NN), dim3(256), 0, stream>>>(X, nf);
    k_node_geo<<<gs(NN), dim3(256), 0, stream>>>(X, nf, lfb);
    gemm(nf, bt_ne, PP(115), nb1, NN, NINP, HID, NINP, HID, 0);          // node_emb
    bn(nb1, PP(9), PP(8), hv_b, NN);                                     // bn_n
    gemm(hv_b, bt_wv, PP(5), hv_a, NN, HID, HID, HID, HID, 0);           // Wv
    float* hV = hv_a; float* hvT = hv_b;

    // ---- edge features + embedding (chunked to cap workspace) ----
    for (int e0 = 0; e0 < NE; e0 += ECH) {
        int c = (NE - e0 < ECH) ? (NE - e0) : ECH;
        k_edge_geo<<<gs(c), dim3(256), 0, stream>>>(X, lfb, src, dst, e0, c, geoE);
        gemm(geoE, bt_ee, PP(11), E1 + (size_t)e0 * HID, c, EINP, HID, EINP, HID, 0);
    }
    bn(E1, PP(7), PP(6), E3, NE);                                        // bn_e
    gemm(E3, bt_we, PP(3), E2, NE, HID, HID, HID, HID, 0);               // We -> h_E

    // ---- GNN layers ----
    for (int L = 0; L < 4; L++) {
        // attention projections
        gemm(hV, btL[L][0], LP(L, 11), qn, NN, HID, HID, HID, HID, 0);   // Wq
        gemm(hV, btL[L][1], LP(L, 8),  kn, NN, HID, HID, HID, HID, 0);   // Wk
        gemm(hV, btL[L][2], LP(L, 12), vn, NN, HID, HID, HID, HID, 0);   // Wv_a
        gemm(E2, btL[L][3], nullptr,   E3, NE, HID, HID, HID, HID, 0);   // We_a
        // segment softmax attention
        k_attn_logits<<<gs((size_t)NE * NHEAD), dim3(256), 0, stream>>>(qn, kn, E3, src, dst, alpha);
        k_fill_u32<<<gs((size_t)NN * NHEAD), dim3(256), 0, stream>>>(mEnc, (size_t)NN * NHEAD, 0x007FFFFFu); // enc(-inf)
        hipMemsetAsync(ssum, 0, (size_t)NN * NHEAD * sizeof(float), stream);
        k_seg_max<<<gs((size_t)NE * NHEAD), dim3(256), 0, stream>>>(alpha, dst, mEnc);
        k_seg_expsum<<<gs((size_t)NE * NHEAD), dim3(256), 0, stream>>>(alpha, dst, mEnc, ssum);
        hipMemsetAsync(dh, 0, (size_t)NN * HID * sizeof(float), stream);
        k_attn_scatter<<<dim3(NE), dim3(HID), 0, stream>>>(vn, E3, alpha, ssum, src, dst, dh);
        k_layernorm<<<dim3(NN), dim3(HID), 0, stream>>>(hV, dh, LP(L, 22), LP(L, 21), hvT);
        { float* t = hV; hV = hvT; hvT = t; }
        // FFN
        gemm(hV, btL[L][4], LP(L, 19), t1, NN, HID, 512, HID, 512, 1);   // relu
        gemm(t1, btL[L][5], LP(L, 20), dh, NN, 512, HID, 512, HID, 0);
        k_layernorm<<<dim3(NN), dim3(HID), 0, stream>>>(hV, dh, LP(L, 24), LP(L, 23), hvT);
        { float* t = hV; hV = hvT; hvT = t; }
        // edge message: gelu(Linear(h_EV)) @ W12, + h_E, batchnorm
        for (int e0 = 0; e0 < NE; e0 += ECH) {
            int c = (NE - e0 < ECH) ? (NE - e0) : ECH;
            k_concat_hEV<<<gs((size_t)c * 384), dim3(256), 0, stream>>>(hV, E2, src, dst, e0, c, hEV);
            gemm(hEV, btL[L][6], LP(L, 6), E4 + (size_t)e0 * HID, c, 384, HID, 384, HID, 2); // gelu
        }
        gemm(E4, btL[L][7], LP(L, 7), E3, NE, HID, HID, HID, HID, 0);    // W12
        k_add<<<gs((size_t)NE * HID), dim3(256), 0, stream>>>(E2, E3, E1, (size_t)NE * HID);
        bn(E1, LP(L, 10), LP(L, 9), E2, NE);                             // new h_E
        // context gating
        hipMemsetAsync(csum, 0, (size_t)NG * HID * sizeof(float), stream);
        k_seg_sum<<<gs((size_t)NN * HID), dim3(256), 0, stream>>>(hV, batch, csum);
        k_div_cnt<<<gs((size_t)NG * HID), dim3(256), 0, stream>>>(csum, cnt, cbuf);
        gemm(cbuf, btL[L][8], LP(L, 15), g1,   NG, HID, HID, HID, HID, 1);  // relu
        gemm(g1,   btL[L][9], LP(L, 16), gate, NG, HID, HID, HID, HID, 3);  // sigmoid
        k_gate_apply<<<gs((size_t)NN * HID), dim3(256), 0, stream>>>(hV, gate, batch);
    }

    // ---- head: elu(fc) + segment mean -> d_out [32 x 128] ----
    gemm(hV, bt_fc, PP(13), nb1, NN, HID, HID, HID, HID, 4);             // elu
    hipMemsetAsync(csum, 0, (size_t)NG * HID * sizeof(float), stream);
    k_seg_sum<<<gs((size_t)NN * HID), dim3(256), 0, stream>>>(nb1, batch, csum);
    k_div_cnt<<<gs((size_t)NG * HID), dim3(256), 0, stream>>>(csum, cnt, (float*)d_out);
}
